// MultiInferLayer_6519760355362
// MI455X (gfx1250) — matrix operand order repaired and checked by reading, not compiled, not run on hardware
//
#include <hip/hip_runtime.h>

typedef __attribute__((ext_vector_type(16))) _Float16     v16h;
typedef __attribute__((ext_vector_type(8)))  float        v8f;
typedef __attribute__((ext_vector_type(8)))  unsigned int v8u;
typedef __attribute__((ext_vector_type(4)))  float        nf4;
typedef __attribute__((ext_vector_type(4)))  unsigned int nu4;

#define BB   4
#define LL   100
#define D2   1536
#define CC   10
#define MROWS (BB * LL * LL)      // 40000
#define KFULL 1566                // D2 + 3C
#define KP    1568                // padded to multiple of 32

// ---------------- workspace layout (bytes, 256-aligned) ----------------
static constexpr size_t OFF_WT     = 0;                                   // f16 [1536][1568]
static constexpr size_t SZ_WT      = (size_t)D2 * KP * 2;
static constexpr size_t OFF_EXT    = OFF_WT + SZ_WT;                      // f16 [40000][32]
static constexpr size_t SZ_EXT     = (size_t)MROWS * 32 * 2;
static constexpr size_t OFF_PROBS  = OFF_EXT + SZ_EXT;                    // f32 [40000][10]
static constexpr size_t SZ_PROBS   = (size_t)MROWS * CC * 4;
static constexpr size_t OFF_POOLED = OFF_PROBS + SZ_PROBS;                // f32 [400][10]
static constexpr size_t SZ_POOLED  = 16384;
static constexpr size_t OFF_FEAT1  = OFF_POOLED + SZ_POOLED;              // f16 [40000][1536]
static constexpr size_t SZ_FEAT    = (size_t)MROWS * D2 * 2;
static constexpr size_t OFF_FEAT2  = OFF_FEAT1 + SZ_FEAT;

// ---------------- W_feat -> f16, N-major, K padded ----------------
__global__ void build_wt_kernel(const float* __restrict__ Wf,   // [1566][1536]
                                _Float16* __restrict__ Wt)      // [1536][1568]
{
    const int t = blockIdx.x * blockDim.x + threadIdx.x;
    if (t >= D2 * KP) return;
    const int n = t / KP, k = t % KP;
    Wt[t] = (k < KFULL) ? (_Float16)Wf[k * D2 + n] : (_Float16)0.0f;
}

// ---------------- cls: logits = softmax(x @ W_cls + b_cls) ----------------
__global__ void __launch_bounds__(256)
cls_softmax_kernel(const float* __restrict__ A32, const _Float16* __restrict__ A16,
                   const float* __restrict__ Wc, const float* __restrict__ bc,
                   float* __restrict__ out /* [MROWS][10] */)
{
    const int wave = (int)((blockIdx.x * blockDim.x + threadIdx.x) >> 5);
    const int lane = threadIdx.x & 31;
    if (wave >= MROWS) return;
    const size_t rowOff = (size_t)wave * D2;

    float acc[CC];
#pragma unroll
    for (int c = 0; c < CC; ++c) acc[c] = 0.0f;

    if (A32) {
        for (int k = lane; k < D2; k += 32) {
            const float x = A32[rowOff + k];
            const float* w = Wc + k * CC;
#pragma unroll
            for (int c = 0; c < CC; ++c) acc[c] += x * w[c];
        }
    } else {
        for (int k = lane; k < D2; k += 32) {
            const float x = (float)A16[rowOff + k];
            const float* w = Wc + k * CC;
#pragma unroll
            for (int c = 0; c < CC; ++c) acc[c] += x * w[c];
        }
    }
#pragma unroll
    for (int c = 0; c < CC; ++c) {
#pragma unroll
        for (int off = 16; off > 0; off >>= 1)
            acc[c] += __shfl_xor(acc[c], off, 32);
    }
    if (lane == 0) {
        float mx = -1e30f;
#pragma unroll
        for (int c = 0; c < CC; ++c) { acc[c] += bc[c]; mx = fmaxf(mx, acc[c]); }
        float s = 0.0f;
#pragma unroll
        for (int c = 0; c < CC; ++c) { acc[c] = expf(acc[c] - mx); s += acc[c]; }
        const float inv = 1.0f / s;
#pragma unroll
        for (int c = 0; c < CC; ++c) out[(size_t)wave * CC + c] = acc[c] * inv;
    }
}

// ---------------- dual-axis masked max pool ----------------
__global__ void __launch_bounds__(256)
pool_kernel(const float* __restrict__ probs, const float* __restrict__ masks,
            float* __restrict__ pooled)
{
    const int wv = (int)((blockIdx.x * blockDim.x + threadIdx.x) >> 5);
    const int lane = threadIdx.x & 31;
    if (wv >= BB * LL) return;
    const int b = wv / LL, l = wv % LL;
    const float ml = masks[b * LL + l];

    float mx[CC];
#pragma unroll
    for (int c = 0; c < CC; ++c) mx[c] = 0.0f;   // masked >= 0; zeros included in max

    for (int j = lane; j < LL; j += 32) {
        const float mj = masks[b * LL + j];
        if (j >= l) {
            const float tv = ml * mj;
            const float* p = probs + ((size_t)(b * LL + l) * LL + j) * CC;
#pragma unroll
            for (int c = 0; c < CC; ++c) mx[c] = fmaxf(mx[c], p[c] * tv);
        }
        if (j <= l) {
            const float tv = ml * mj;
            const float* p = probs + ((size_t)(b * LL + j) * LL + l) * CC;
#pragma unroll
            for (int c = 0; c < CC; ++c) mx[c] = fmaxf(mx[c], p[c] * tv);
        }
    }
#pragma unroll
    for (int c = 0; c < CC; ++c) {
#pragma unroll
        for (int off = 16; off > 0; off >>= 1)
            mx[c] = fmaxf(mx[c], __shfl_xor(mx[c], off, 32));
    }
    if (lane == 0) {
#pragma unroll
        for (int c = 0; c < CC; ++c) pooled[wv * CC + c] = mx[c];
    }
}

// ---------------- pack 30 concat-tail values per row, f16, K-padded to 32 ----
__global__ void pack_ext_kernel(const float* __restrict__ probs,
                                const float* __restrict__ pooled,
                                _Float16* __restrict__ ext /* [MROWS][32] */)
{
    const int t = blockIdx.x * blockDim.x + threadIdx.x;
    if (t >= MROWS * 32) return;
    const int row = t >> 5, s = t & 31;
    const int b = row / (LL * LL);
    const int i = (row / LL) % LL;
    const int j = row % LL;
    float v = 0.0f;
    if (s < 10)      v = pooled[(b * LL + i) * CC + s];
    else if (s < 20) v = pooled[(b * LL + j) * CC + (s - 10)];
    else if (s < 30) v = probs[(size_t)row * CC + (s - 20)];
    ext[t] = (_Float16)v;
}

// ---------------- WMMA GEMM, double-buffered LDS pipeline ----------------
// block: 256 thr = 8 waves (2 in M x 4 in N). Tile: M=64, N=256, K-step=32.
template<bool F32SRC>
__global__ void __launch_bounds__(256)
gemm_feat_kernel(const float* __restrict__ A32,      // [M][1536] fp32 (hop 0)
                 const _Float16* __restrict__ A16,   // [M][1536] f16 (hop 1)
                 const _Float16* __restrict__ Aext,  // [M][32]  f16 K-tail
                 const _Float16* __restrict__ Wt,    // [1536][KP] N-major f16
                 const float* __restrict__ bias,     // [1536]
                 _Float16* __restrict__ Out)         // [M][1536] f16
{
    __shared__ _Float16 As[2][64][40];   // +8 pad keeps 128b frag reads conflict-free
    __shared__ _Float16 Bs[2][256][40];

    const int m0   = blockIdx.x * 64;
    const int n0   = blockIdx.y * 256;
    const int tid  = threadIdx.x;
    const int lane = tid & 31;
    const int w    = tid >> 5;
    const int wm   = w & 1;        // rows [wm*32, +32)
    const int wn   = w >> 1;       // cols [wn*64, +64)
    const int hl   = lane >> 4;
    const int l16  = lane & 15;

    v8f acc[2][4] = {};

    // tile-fill indices: A = 64 rows x 4 chunks of 8 halves; B = 1 row (32 halves) / thread
    const int arow = tid >> 2, aq = tid & 3;
    const size_t aMainOff = (size_t)(m0 + arow) * D2 + aq * 8;   // + kb
    const size_t aExtOff  = (size_t)(m0 + arow) * 32 + aq * 8;
    const _Float16* bBase = Wt + (size_t)(n0 + tid) * KP;        // + kb

    // prefetch registers
    nf4 pf0 = {}, pf1 = {};
    nu4 pa = {}, pb0, pb1, pb2, pb3;

    auto issue_loads = [&](int kb) {
        if (kb < D2) {
            if constexpr (F32SRC) {
                const nf4* s = (const nf4*)(A32 + aMainOff + kb);
                pf0 = __builtin_nontemporal_load(s);         // streaming input, spare L2
                pf1 = __builtin_nontemporal_load(s + 1);
            } else {
                pa = *(const nu4*)(A16 + aMainOff + kb);
            }
        } else {
            pa = *(const nu4*)(Aext + aExtOff);
        }
        const _Float16* bs = bBase + kb;
        pb0 = *(const nu4*)(bs);
        pb1 = *(const nu4*)(bs + 8);
        pb2 = *(const nu4*)(bs + 16);
        pb3 = *(const nu4*)(bs + 24);
    };

    auto store_tile = [&](int p, bool isExt) {
        _Float16* ad = &As[p][arow][aq * 8];
        if (F32SRC && !isExt) {
            ad[0] = (_Float16)pf0.x; ad[1] = (_Float16)pf0.y;
            ad[2] = (_Float16)pf0.z; ad[3] = (_Float16)pf0.w;
            ad[4] = (_Float16)pf1.x; ad[5] = (_Float16)pf1.y;
            ad[6] = (_Float16)pf1.z; ad[7] = (_Float16)pf1.w;
        } else {
            *(nu4*)ad = pa;
        }
        _Float16* bd = &Bs[p][tid][0];
        *(nu4*)(bd)      = pb0;
        *(nu4*)(bd + 8)  = pb1;
        *(nu4*)(bd + 16) = pb2;
        *(nu4*)(bd + 24) = pb3;
    };

    auto compute = [&](int p) {
        // A frags: ISA 16-bit A layout (lanes0-15 K{0..7,16..23}; lanes16-31 K{8..15,24..31})
        v16h afrag[2];
#pragma unroll
        for (int s = 0; s < 2; ++s) {
            const int r = wm * 32 + s * 16 + l16;
            v8u au;
#pragma unroll
            for (int v = 0; v < 8; ++v) {
                const int k0 = 2 * v + ((v >= 4) ? 8 : 0) + 8 * hl;
                au[v] = *(const unsigned int*)&As[p][r][k0];
            }
            afrag[s] = __builtin_bit_cast(v16h, au);
        }
        // B frags: column per lane; lanes0-15 K{0..7,16..23}; lanes16-31 K{8..15,24..31}
#pragma unroll
        for (int t = 0; t < 4; ++t) {
            const int nr = wn * 64 + t * 16 + l16;
            v8u bu;
#pragma unroll
            for (int v = 0; v < 8; ++v) {
                const int k0 = 2 * v + ((v >= 4) ? 8 : 0) + 8 * hl;
                bu[v] = *(const unsigned int*)&Bs[p][nr][k0];
            }
            const v16h bfrag = __builtin_bit_cast(v16h, bu);
#pragma unroll
            for (int s = 0; s < 2; ++s) {
                acc[s][t] = __builtin_amdgcn_wmma_f32_16x16x32_f16(
                    false, afrag[s], false, bfrag, (short)0, acc[s][t], false, false);
            }
        }
    };

    // ---- software pipeline: load(k+1) || compute(k) ----
    issue_loads(0);
    store_tile(0, false);
    __syncthreads();
    int p = 0;
    for (int kb = 32; kb < KP; kb += 32) {
        issue_loads(kb);                 // global loads in flight during compute
        compute(p);
        store_tile(p ^ 1, kb >= D2);     // s_wait_loadcnt lands here, after the WMMAs
        __syncthreads();                 // one barrier per K-step
        p ^= 1;
    }
    compute(p);

    // ---- epilogue: +bias, f16 store (C layout: VGPR g -> m = g + 8*half, n = lane%16) ----
#pragma unroll
    for (int t = 0; t < 4; ++t) {
        const int ncol = n0 + wn * 64 + t * 16 + l16;
        const float bv = bias[ncol];
#pragma unroll
        for (int s = 0; s < 2; ++s) {
#pragma unroll
            for (int g = 0; g < 8; ++g) {
                const int grow = m0 + wm * 32 + s * 16 + g + 8 * hl;
                Out[(size_t)grow * D2 + ncol] = (_Float16)(acc[s][t][g] + bv);
            }
        }
    }
}

// ---------------- launcher ----------------
extern "C" void kernel_launch(void* const* d_in, const int* in_sizes, int n_in,
                              void* d_out, int out_size, void* d_ws, size_t ws_size,
                              hipStream_t stream)
{
    const float* feat  = (const float*)d_in[0];   // [4,100,100,1536]
    const float* masks = (const float*)d_in[1];   // [4,100]
    const float* Wc    = (const float*)d_in[2];   // [1536,10]
    const float* bc    = (const float*)d_in[3];   // [10]
    const float* Wf    = (const float*)d_in[4];   // [1566,1536]
    const float* bf    = (const float*)d_in[5];   // [1536]
    float* out = (float*)d_out;                    // [4,100,100,10]

    char* ws = (char*)d_ws;
    _Float16* Wt     = (_Float16*)(ws + OFF_WT);
    _Float16* ext    = (_Float16*)(ws + OFF_EXT);
    float*    probs  = (float*)(ws + OFF_PROBS);
    float*    pooled = (float*)(ws + OFF_POOLED);
    _Float16* feat1  = (_Float16*)(ws + OFF_FEAT1);
    _Float16* feat2  = (_Float16*)(ws + OFF_FEAT2);

    const dim3 gemmGrid(MROWS / 64, D2 / 256);   // 625 x 6

    build_wt_kernel<<<(D2 * KP + 255) / 256, 256, 0, stream>>>(Wf, Wt);

    cls_softmax_kernel<<<MROWS / 8, 256, 0, stream>>>(feat, nullptr, Wc, bc, probs);

    // ---- hop 0 ----
    pool_kernel<<<(BB * LL * 32 + 255) / 256, 256, 0, stream>>>(probs, masks, pooled);
    pack_ext_kernel<<<(MROWS * 32 + 255) / 256, 256, 0, stream>>>(probs, pooled, ext);
    gemm_feat_kernel<true><<<gemmGrid, 256, 0, stream>>>(feat, nullptr, ext, Wt, bf, feat1);
    cls_softmax_kernel<<<MROWS / 8, 256, 0, stream>>>(nullptr, feat1, Wc, bc, probs);

    // ---- hop 1 ----
    pool_kernel<<<(BB * LL * 32 + 255) / 256, 256, 0, stream>>>(probs, masks, pooled);
    pack_ext_kernel<<<(MROWS * 32 + 255) / 256, 256, 0, stream>>>(probs, pooled, ext);
    gemm_feat_kernel<false><<<gemmGrid, 256, 0, stream>>>(nullptr, feat1, ext, Wt, bf, feat2);
    cls_softmax_kernel<<<MROWS / 8, 256, 0, stream>>>(nullptr, feat2, Wc, bc, out);
}
